// MoeModel_8083128451723
// MI455X (gfx1250) — matrix operand order repaired and checked by reading, not compiled, not run on hardware
//
#include <hip/hip_runtime.h>

// ---------------- model dims ----------------
#define B_    16384
#define OBS_  256
#define LAT_  256
#define H_    256
#define OUT_  12
#define E_    16
#define CNN_LAT_ 192

typedef unsigned short u16;
typedef __attribute__((ext_vector_type(16))) __bf16 v16bf;
typedef __attribute__((ext_vector_type(8)))  __bf16 v8bf;
typedef __attribute__((ext_vector_type(8)))  float  v8f;
typedef __attribute__((ext_vector_type(4)))  int    v4i;

typedef __attribute__((address_space(1))) v4i* gv4i_p;  // global b128
typedef __attribute__((address_space(3))) v4i* lv4i_p;  // LDS b128

union V16U { v16bf v; v8bf h[2]; };

#if __has_builtin(__builtin_amdgcn_global_load_async_to_lds_b128)
#define HAVE_ASYNC_LDS 1
#endif

__device__ __forceinline__ u16 f2bf(float f) {
  unsigned u = __float_as_uint(f);
  unsigned r = u + 0x7FFFu + ((u >> 16) & 1u);   // round-to-nearest-even
  return (u16)(r >> 16);
}
__device__ __forceinline__ float bf2f(u16 u) {
  return __uint_as_float(((unsigned)u) << 16);
}
__device__ __forceinline__ float elu(float x) {
  return x > 0.f ? x : (expf(x) - 1.f);
}

__device__ __forceinline__ void async_wait0() {
#if __has_builtin(__builtin_amdgcn_s_wait_asynccnt)
  __builtin_amdgcn_s_wait_asynccnt(0);
#else
  asm volatile("s_wait_asynccnt 0x0" ::: "memory");
#endif
}

// Stage `elems` u16 (multiple of 8) from global to LDS.
// Uses GLOBAL_LOAD_ASYNC_TO_LDS_B128 (ASYNCcnt path) when available.
__device__ __forceinline__ void stage_tile(u16* lds, const u16* gsrc, int elems,
                                           int tid, int nthreads) {
#ifdef HAVE_ASYNC_LDS
  for (int i = tid * 8; i < elems; i += nthreads * 8)
    __builtin_amdgcn_global_load_async_to_lds_b128(
        (gv4i_p)(gsrc + i), (lv4i_p)(lds + i), 0, 0);
  async_wait0();
#else
  for (int i = tid; i < elems; i += nthreads) lds[i] = gsrc[i];
#endif
}

// A fragment: 16x32 bf16, row-major source [16][ldk] (LDS or global).
// Layout used here: lane m=l&15; g=(l>>4)*8; K covered: 2g..2g+15 (contiguous, matches B).
__device__ __forceinline__ v16bf load_A(const u16* base, int ldk, int k0, int lane) {
  int m = lane & 15;
  int g = (lane >> 4) * 8;
  V16U r;
  r.h[0] = *(const v8bf*)(base + m * ldk + k0 + 2 * g);
  r.h[1] = *(const v8bf*)(base + m * ldk + k0 + 2 * g + 8);
  return r.v;
}

// B fragment: 32x16 bf16 from pre-swizzled tile of 512 bf16:
// element (l,t): K=(l>>4)*16+t, N=l&15  -> lane reads 32 contiguous bytes.
__device__ __forceinline__ v16bf load_B(const u16* tile, int lane) {
  V16U r;
  r.h[0] = *(const v8bf*)(tile + lane * 16);
  r.h[1] = *(const v8bf*)(tile + lane * 16 + 8);
  return r.v;
}

__device__ __forceinline__ v8f wmma_bf16(v16bf a, v16bf b, v8f c) {
  return __builtin_amdgcn_wmma_f32_16x16x32_bf16(false, a, false, b, (short)0, c,
                                                 false, false);
}

// ---------------------------------------------------------------------------
// Weight swizzle: f32 [nmat][K][Nsrc] -> bf16 tiles [nmat][Npad/16][K/32][512]
// ---------------------------------------------------------------------------
__global__ void __launch_bounds__(256)
swz_kernel(const float* __restrict__ src, u16* __restrict__ dst,
           int K, int Nsrc, int Npad, int nmat) {
  long long i = (long long)blockIdx.x * 256 + threadIdx.x;
  long long total = (long long)nmat * (Npad / 16) * (K / 32) * 512;
  if (i >= total) return;
  int t = (int)(i & 15);
  int l = (int)((i >> 4) & 31);
  long long tile = i >> 9;
  int ksteps = K / 32, ntiles = Npad / 16;
  int kstep = (int)(tile % ksteps);
  long long rest = tile / ksteps;
  int ntile = (int)(rest % ntiles);
  int mat = (int)(rest / ntiles);
  int n = ntile * 16 + (l & 15);
  int k = kstep * 32 + (l >> 4) * 16 + t;
  float v = (n < Nsrc) ? src[((long long)mat * K + k) * Nsrc + n] : 0.f;
  dst[i] = f2bf(v);
}

// ---------------------------------------------------------------------------
// Fused conv1+conv2 (stride-2 SAME, elu), 4 samples/block, bf16 out [B][2048]
// flatten order c*64 + y*8 + x (matches reshape of [32,8,8]).
// ---------------------------------------------------------------------------
#define CONV_S 4
__global__ void __launch_bounds__(256)
conv_kernel(const float* __restrict__ cam,
            const float* __restrict__ w1, const float* __restrict__ b1,
            const float* __restrict__ w2, const float* __restrict__ b2,
            u16* __restrict__ conv2out) {
  __shared__ u16 c1[CONV_S * 16 * 16 * 16];   // [s][c][y][x] bf16, 32 KB
  __shared__ float sw2[32 * 16 * 9];          // 18 KB
  __shared__ float sw1[16 * 9];
  __shared__ float sb1[16], sb2[32];
  int tid = threadIdx.x;
  long long b0 = (long long)blockIdx.x * CONV_S;
  for (int i = tid; i < 32 * 16 * 9; i += 256) sw2[i] = w2[i];
  for (int i = tid; i < 16 * 9; i += 256) sw1[i] = w1[i];
  if (tid < 16) sb1[tid] = b1[tid];
  if (tid < 32) sb2[tid] = b2[tid];
  __syncthreads();
  // conv1: [s][16][16][16]
  for (int i = tid; i < CONV_S * 16 * 256; i += 256) {
    int x = i & 15, y = (i >> 4) & 15, c = (i >> 8) & 15, s = i >> 12;
    const float* img = cam + (b0 + s) * 1024;
    float acc = sb1[c];
    #pragma unroll
    for (int ky = 0; ky < 3; ky++) {
      int iy = 2 * y + ky;
      if (iy >= 32) continue;                  // SAME pad: lo=0, hi=1
      #pragma unroll
      for (int kx = 0; kx < 3; kx++) {
        int ix = 2 * x + kx;
        if (ix >= 32) continue;
        acc += img[iy * 32 + ix] * sw1[c * 9 + ky * 3 + kx];
      }
    }
    c1[i] = f2bf(elu(acc));
  }
  __syncthreads();
  // conv2: [s][32][8][8]
  for (int i = tid; i < CONV_S * 32 * 64; i += 256) {
    int x = i & 7, y = (i >> 3) & 7, c = (i >> 6) & 31, s = i >> 11;
    float acc = sb2[c];
    for (int ic = 0; ic < 16; ic++) {
      const u16* pin = c1 + ((s * 16 + ic) << 8);
      const float* pw = sw2 + (c * 16 + ic) * 9;
      #pragma unroll
      for (int ky = 0; ky < 3; ky++) {
        int iy = 2 * y + ky;
        if (iy >= 16) continue;
        #pragma unroll
        for (int kx = 0; kx < 3; kx++) {
          int ix = 2 * x + kx;
          if (ix >= 16) continue;
          acc += bf2f(pin[iy * 16 + ix]) * pw[ky * 3 + kx];
        }
      }
    }
    conv2out[(b0 + s) * 2048 + c * 64 + y * 8 + x] = f2bf(elu(acc));
  }
}

// ---------------------------------------------------------------------------
// CNN linear [B,2048]x[2048,192] via WMMA + bias + LayerNorm -> latent[:,64:256]
// 16 rows/block, 8 waves, 12 N-tiles, 64 K-steps. A-tile staged once in LDS
// (async-to-LDS), then re-read 12x from LDS instead of 12x from L2.
// ---------------------------------------------------------------------------
__global__ void __launch_bounds__(256)
cnn_lin_kernel(const u16* __restrict__ conv2out, const u16* __restrict__ wB,
               const float* __restrict__ bias, const float* __restrict__ g,
               const float* __restrict__ bb, u16* __restrict__ lat) {
  __shared__ u16 sA[16 * 2048];        // 64 KB
  __shared__ float ybuf[16 * 192];
  __shared__ float mv[32];
  int tid = threadIdx.x, lane = tid & 31, wave = tid >> 5;
  long long b0 = (long long)blockIdx.x * 16;
  stage_tile(sA, conv2out + b0 * 2048, 16 * 2048, tid, 256);
  __syncthreads();
  for (int ti = wave; ti < 12; ti += 8) {
    v8f acc = {};
    const u16* bt = wB + ti * 64 * 512;
    for (int ks = 0; ks < 64; ks++)
      acc = wmma_bf16(load_A(sA, 2048, ks * 32, lane), load_B(bt + ks * 512, lane), acc);
    int n = ti * 16 + (lane & 15);
    int mb = (lane >> 4) * 8;
    float bv = bias[n];
    #pragma unroll
    for (int r = 0; r < 8; r++) ybuf[(mb + r) * 192 + n] = acc[r] + bv;
  }
  __syncthreads();
  if (tid < 16) {
    float m = 0.f;
    for (int c = 0; c < 192; c++) m += ybuf[tid * 192 + c];
    m /= 192.f;
    float v = 0.f;
    for (int c = 0; c < 192; c++) { float d = ybuf[tid * 192 + c] - m; v += d * d; }
    v /= 192.f;
    mv[tid * 2] = m;
    mv[tid * 2 + 1] = rsqrtf(v + 1e-5f);
  }
  __syncthreads();
  for (int i = tid; i < 16 * 192; i += 256) {
    int r = i / 192, c = i % 192;
    lat[(b0 + r) * 256 + 64 + c] = f2bf((ybuf[i] - mv[r * 2]) * mv[r * 2 + 1] * g[c] + bb[c]);
  }
}

// ---------------------------------------------------------------------------
// Proprio encoder 256->128(elu)->64 + LayerNorm -> latent[:,0:64] (WMMA)
// ---------------------------------------------------------------------------
__global__ void __launch_bounds__(256)
enc_kernel(const float* __restrict__ proprio,
           const u16* __restrict__ w1B, const float* __restrict__ b1,
           const u16* __restrict__ w2B, const float* __restrict__ b2,
           const float* __restrict__ g, const float* __restrict__ bb,
           u16* __restrict__ lat) {
  __shared__ u16 aP[16 * 256];
  __shared__ u16 h1[16 * 128];
  __shared__ float ybuf[16 * 64];
  __shared__ float mv[32];
  int tid = threadIdx.x, lane = tid & 31, wave = tid >> 5;
  long long b0 = (long long)blockIdx.x * 16;
  for (int i = tid; i < 16 * 256; i += 256) aP[i] = f2bf(proprio[b0 * 256 + i]);
  __syncthreads();
  {  // layer1: 8 tiles of 16 cols, one per wave; K=256
    v8f acc = {};
    const u16* bt = w1B + wave * 8 * 512;
    #pragma unroll
    for (int ks = 0; ks < 8; ks++)
      acc = wmma_bf16(load_A(aP, 256, ks * 32, lane), load_B(bt + ks * 512, lane), acc);
    int n = wave * 16 + (lane & 15), mb = (lane >> 4) * 8;
    float bv = b1[n];
    #pragma unroll
    for (int r = 0; r < 8; r++) h1[(mb + r) * 128 + n] = f2bf(elu(acc[r] + bv));
  }
  __syncthreads();
  if (wave < 4) {  // layer2: 4 tiles; K=128
    v8f acc = {};
    const u16* bt = w2B + wave * 4 * 512;
    #pragma unroll
    for (int ks = 0; ks < 4; ks++)
      acc = wmma_bf16(load_A(h1, 128, ks * 32, lane), load_B(bt + ks * 512, lane), acc);
    int n = wave * 16 + (lane & 15), mb = (lane >> 4) * 8;
    float bv = b2[n];
    #pragma unroll
    for (int r = 0; r < 8; r++) ybuf[(mb + r) * 64 + n] = acc[r] + bv;
  }
  __syncthreads();
  if (tid < 16) {
    float m = 0.f;
    for (int c = 0; c < 64; c++) m += ybuf[tid * 64 + c];
    m /= 64.f;
    float v = 0.f;
    for (int c = 0; c < 64; c++) { float d = ybuf[tid * 64 + c] - m; v += d * d; }
    v /= 64.f;
    mv[tid * 2] = m;
    mv[tid * 2 + 1] = rsqrtf(v + 1e-5f);
  }
  __syncthreads();
  for (int i = tid; i < 16 * 64; i += 256) {
    int r = i >> 6, c = i & 63;
    lat[(b0 + r) * 256 + c] = f2bf((ybuf[i] - mv[r * 2]) * mv[r * 2 + 1] * g[c] + bb[c]);
  }
}

// ---------------------------------------------------------------------------
// Router 256->128(elu)->64(elu)->16 + softmax -> gate [B,16] (WMMA)
// ---------------------------------------------------------------------------
__global__ void __launch_bounds__(256)
router_kernel(const u16* __restrict__ lat,
              const u16* __restrict__ w1B, const float* __restrict__ b1,
              const u16* __restrict__ w2B, const float* __restrict__ b2,
              const u16* __restrict__ w3B, const float* __restrict__ b3,
              float* __restrict__ gate) {
  __shared__ u16 aL[16 * 256];
  __shared__ u16 r1[16 * 128];
  __shared__ u16 r2[16 * 64];
  __shared__ float logits[16 * 16];
  int tid = threadIdx.x, lane = tid & 31, wave = tid >> 5;
  long long b0 = (long long)blockIdx.x * 16;
  stage_tile(aL, lat + b0 * 256, 16 * 256, tid, 256);
  __syncthreads();
  {
    v8f acc = {};
    const u16* bt = w1B + wave * 8 * 512;
    #pragma unroll
    for (int ks = 0; ks < 8; ks++)
      acc = wmma_bf16(load_A(aL, 256, ks * 32, lane), load_B(bt + ks * 512, lane), acc);
    int n = wave * 16 + (lane & 15), mb = (lane >> 4) * 8;
    float bv = b1[n];
    #pragma unroll
    for (int r = 0; r < 8; r++) r1[(mb + r) * 128 + n] = f2bf(elu(acc[r] + bv));
  }
  __syncthreads();
  if (wave < 4) {
    v8f acc = {};
    const u16* bt = w2B + wave * 4 * 512;
    #pragma unroll
    for (int ks = 0; ks < 4; ks++)
      acc = wmma_bf16(load_A(r1, 128, ks * 32, lane), load_B(bt + ks * 512, lane), acc);
    int n = wave * 16 + (lane & 15), mb = (lane >> 4) * 8;
    float bv = b2[n];
    #pragma unroll
    for (int r = 0; r < 8; r++) r2[(mb + r) * 64 + n] = f2bf(elu(acc[r] + bv));
  }
  __syncthreads();
  if (wave == 0) {
    v8f acc = {};
    #pragma unroll
    for (int ks = 0; ks < 2; ks++)
      acc = wmma_bf16(load_A(r2, 64, ks * 32, lane), load_B(w3B + ks * 512, lane), acc);
    int n = lane & 15, mb = (lane >> 4) * 8;
    float bv = b3[n];
    #pragma unroll
    for (int r = 0; r < 8; r++) logits[(mb + r) * 16 + n] = acc[r] + bv;
  }
  __syncthreads();
  if (tid < 16) {
    float mx = -1e30f;
    for (int e = 0; e < 16; e++) mx = fmaxf(mx, logits[tid * 16 + e]);
    float s = 0.f;
    for (int e = 0; e < 16; e++) s += expf(logits[tid * 16 + e] - mx);
    float inv = 1.f / s;
    for (int e = 0; e < 16; e++)
      gate[(b0 + tid) * 16 + e] = expf(logits[tid * 16 + e] - mx) * inv;
  }
}

// ---------------------------------------------------------------------------
// Experts: one block = 16 batch rows x all 16 experts (A-tile reused in LDS).
// Each hidden layer: 16 N-tiles over 8 waves. Both of a wave's tiles are
// computed together: one A-fragment feeds two independent WMMA chains.
// ---------------------------------------------------------------------------
__device__ __forceinline__ void layer256(const u16* A, const u16* W,
                                         const float* bias, u16* Dst,
                                         int lane, int wave) {
  v8f acc0 = {}, acc1 = {};
  const u16* bt0 = W + (wave    ) * 8 * 512;
  const u16* bt1 = W + (wave + 8) * 8 * 512;
  #pragma unroll
  for (int ks = 0; ks < 8; ks++) {
    v16bf a = load_A(A, 256, ks * 32, lane);   // shared A fragment
    acc0 = wmma_bf16(a, load_B(bt0 + ks * 512, lane), acc0);
    acc1 = wmma_bf16(a, load_B(bt1 + ks * 512, lane), acc1);
  }
  int nn = lane & 15, mb = (lane >> 4) * 8;
  int n0 = wave * 16 + nn, n1 = (wave + 8) * 16 + nn;
  float bv0 = bias[n0], bv1 = bias[n1];
  #pragma unroll
  for (int r = 0; r < 8; r++) {
    Dst[(mb + r) * 256 + n0] = f2bf(elu(acc0[r] + bv0));
    Dst[(mb + r) * 256 + n1] = f2bf(elu(acc1[r] + bv1));
  }
}

__global__ void __launch_bounds__(256)
expert_kernel(const u16* __restrict__ lat, const float* __restrict__ gate,
              const u16* __restrict__ w123,  // [3][E][16][8][512]
              const u16* __restrict__ w4,    // [E][8][512] (N padded 12->16)
              const float* __restrict__ eb1, const float* __restrict__ eb2,
              const float* __restrict__ eb3, const float* __restrict__ eb4,
              float* __restrict__ out) {
  __shared__ u16 aL[16 * 256];
  __shared__ u16 hA[16 * 256];
  __shared__ u16 hB[16 * 256];
  __shared__ float sgate[16 * 16];
  __shared__ float outacc[16 * 16];
  int tid = threadIdx.x, lane = tid & 31, wave = tid >> 5;
  long long b0 = (long long)blockIdx.x * 16;
  stage_tile(aL, lat + b0 * 256, 16 * 256, tid, 256);
  for (int i = tid; i < 256; i += 256) {
    outacc[i] = 0.f;
    sgate[i] = gate[b0 * 16 + i];
  }
  __syncthreads();
  const int MATSZ = 16 * 8 * 512;  // one 256x256 swizzled matrix
  for (int e = 0; e < E_; e++) {
    const u16* W1 = w123 + (long long)(0 * E_ + e) * MATSZ;
    const u16* W2 = w123 + (long long)(1 * E_ + e) * MATSZ;
    const u16* W3 = w123 + (long long)(2 * E_ + e) * MATSZ;
    if (e + 1 < E_)  // warm next expert's first weight tiles (global_prefetch_b8)
      __builtin_prefetch(w123 + (long long)(e + 1) * MATSZ + (long long)lane * 64, 0, 1);
    layer256(aL, W1, eb1 + e * H_, hA, lane, wave);
    __syncthreads();
    layer256(hA, W2, eb2 + e * H_, hB, lane, wave);
    __syncthreads();
    layer256(hB, W3, eb3 + e * H_, hA, lane, wave);
    __syncthreads();
    if (wave == 0) {  // head: 256 -> 12 (padded 16), gate-weighted accumulate
      v8f acc = {};
      const u16* W4 = w4 + e * 8 * 512;
      #pragma unroll
      for (int ks = 0; ks < 8; ks++)
        acc = wmma_bf16(load_A(hA, 256, ks * 32, lane), load_B(W4 + ks * 512, lane), acc);
      int n = lane & 15, mb = (lane >> 4) * 8;
      if (n < OUT_) {
        float bv = eb4[e * OUT_ + n];
        #pragma unroll
        for (int r = 0; r < 8; r++) {
          int m = mb + r;
          outacc[m * 16 + n] += sgate[m * 16 + e] * (acc[r] + bv);
        }
      }
    }
    __syncthreads();
  }
  for (int i = tid; i < 16 * OUT_; i += 256) {
    int r = i / OUT_, c = i % OUT_;
    out[(b0 + r) * OUT_ + c] = outacc[r * 16 + c];
  }
}

// ---------------------------------------------------------------------------
// Host side
// ---------------------------------------------------------------------------
extern "C" void kernel_launch(void* const* d_in, const int* in_sizes, int n_in,
                              void* d_out, int out_size, void* d_ws, size_t ws_size,
                              hipStream_t stream) {
  const float* proprio   = (const float*)d_in[0];
  const float* camera    = (const float*)d_in[1];
  const float* enc_w1    = (const float*)d_in[2];
  const float* enc_b1    = (const float*)d_in[3];
  const float* enc_w2    = (const float*)d_in[4];
  const float* enc_b2    = (const float*)d_in[5];
  const float* enc_ln_g  = (const float*)d_in[6];
  const float* enc_ln_b  = (const float*)d_in[7];
  const float* conv1_w   = (const float*)d_in[8];
  const float* conv1_b   = (const float*)d_in[9];
  const float* conv2_w   = (const float*)d_in[10];
  const float* conv2_b   = (const float*)d_in[11];
  const float* cnn_lin_w = (const float*)d_in[12];
  const float* cnn_lin_b = (const float*)d_in[13];
  const float* cnn_ln_g  = (const float*)d_in[14];
  const float* cnn_ln_b  = (const float*)d_in[15];
  const float* rw1 = (const float*)d_in[16];
  const float* rb1 = (const float*)d_in[17];
  const float* rw2 = (const float*)d_in[18];
  const float* rb2 = (const float*)d_in[19];
  const float* rw3 = (const float*)d_in[20];
  const float* rb3 = (const float*)d_in[21];
  const float* ew1 = (const float*)d_in[22];
  const float* eb1 = (const float*)d_in[23];
  const float* ew2 = (const float*)d_in[24];
  const float* eb2 = (const float*)d_in[25];
  const float* ew3 = (const float*)d_in[26];
  const float* eb3 = (const float*)d_in[27];
  const float* ew4 = (const float*)d_in[28];
  const float* eb4 = (const float*)d_in[29];
  float* out = (float*)d_out;

  // ---- workspace carve-up (~84 MB total) ----
  char* ws = (char*)d_ws;
  auto carve = [&](size_t bytes) -> char* {
    char* p = ws;
    ws += (bytes + 255) & ~(size_t)255;
    return p;
  };
  u16*   wLat   = (u16*)carve((size_t)B_ * 256 * 2);          // latent bf16
  float* wGate  = (float*)carve((size_t)B_ * 16 * 4);         // gate f32
  u16*   wConv2 = (u16*)carve((size_t)B_ * 2048 * 2);         // conv2 act bf16
  u16*   wE123  = (u16*)carve((size_t)3 * E_ * 16 * 8 * 512 * 2);
  u16*   wE4    = (u16*)carve((size_t)E_ * 8 * 512 * 2);
  u16*   wCnnL  = (u16*)carve((size_t)12 * 64 * 512 * 2);
  u16*   wEnc1  = (u16*)carve((size_t)8 * 8 * 512 * 2);
  u16*   wEnc2  = (u16*)carve((size_t)4 * 4 * 512 * 2);
  u16*   wR1    = (u16*)carve((size_t)8 * 8 * 512 * 2);
  u16*   wR2    = (u16*)carve((size_t)4 * 4 * 512 * 2);
  u16*   wR3    = (u16*)carve((size_t)1 * 2 * 512 * 2);

  auto swz = [&](const float* src, u16* dst, int K, int Nsrc, int Npad, int nmat) {
    long long total = (long long)nmat * (Npad / 16) * (K / 32) * 512;
    int blocks = (int)((total + 255) / 256);
    swz_kernel<<<blocks, 256, 0, stream>>>(src, dst, K, Nsrc, Npad, nmat);
  };
  const int MATSZ = 16 * 8 * 512;
  swz(ew1, wE123 + 0 * E_ * MATSZ, 256, 256, 256, E_);
  swz(ew2, wE123 + 1 * E_ * MATSZ, 256, 256, 256, E_);
  swz(ew3, wE123 + 2 * E_ * MATSZ, 256, 256, 256, E_);
  swz(ew4, wE4, 256, 12, 16, E_);
  swz(cnn_lin_w, wCnnL, 2048, 192, 192, 1);
  swz(enc_w1, wEnc1, 256, 128, 128, 1);
  swz(enc_w2, wEnc2, 128, 64, 64, 1);
  swz(rw1, wR1, 256, 128, 128, 1);
  swz(rw2, wR2, 128, 64, 64, 1);
  swz(rw3, wR3, 64, 16, 16, 1);

  conv_kernel<<<B_ / CONV_S, 256, 0, stream>>>(camera, conv1_w, conv1_b,
                                               conv2_w, conv2_b, wConv2);
  cnn_lin_kernel<<<B_ / 16, 256, 0, stream>>>(wConv2, wCnnL, cnn_lin_b,
                                              cnn_ln_g, cnn_ln_b, wLat);
  enc_kernel<<<B_ / 16, 256, 0, stream>>>(proprio, wEnc1, enc_b1, wEnc2, enc_b2,
                                          enc_ln_g, enc_ln_b, wLat);
  router_kernel<<<B_ / 16, 256, 0, stream>>>(wLat, wR1, rb1, wR2, rb2, wR3, rb3,
                                             wGate);
  expert_kernel<<<B_ / 16, 256, 0, stream>>>(wLat, wGate, wE123, wE4,
                                             eb1, eb2, eb3, eb4, out);
}
